// MultiHeadSelfAttention_17222818857197
// MI455X (gfx1250) — matrix operand order repaired and checked by reading, not compiled, not run on hardware
//
#include <hip/hip_runtime.h>

// MI455X / gfx1250 multi-head self-attention forward.
// bf16 WMMA (v_wmma_f32_16x16x32_bf16) for all matmuls, f32 accumulation,
// flash-style streaming softmax (exp2 domain), async global->LDS staging
// (ASYNCcnt) with double-buffered LDS so DMA overlaps the matrix pipe.

typedef __attribute__((ext_vector_type(16))) __bf16 v16bf;
typedef __attribute__((ext_vector_type(8)))  float  v8f;

#define WMMA_BF16(a, b, c) \
  __builtin_amdgcn_wmma_f32_16x16x32_bf16(false, (a), false, (b), (short)0, (c), false, false)

#if __has_builtin(__builtin_amdgcn_exp2f)
#define EXP2F __builtin_amdgcn_exp2f
#else
#define EXP2F __builtin_exp2f
#endif

union FragU {
  v16bf v;
  unsigned int u[8];
};

__device__ __forceinline__ unsigned short f32_to_bf16(float f) {
  unsigned int u = __float_as_uint(f);
  u += 0x7FFFu + ((u >> 16) & 1u);  // round-to-nearest-even
  return (unsigned short)(u >> 16);
}
__device__ __forceinline__ float bf16_to_f32(unsigned short h) {
  return __uint_as_float((unsigned int)h << 16);
}
__device__ __forceinline__ unsigned int pack_bf16x2(float lo, float hi) {
  return (unsigned int)f32_to_bf16(lo) | ((unsigned int)f32_to_bf16(hi) << 16);
}

// Async 16B global->LDS copy (GLOBAL_LOAD_ASYNC_TO_LDS_B128, ASYNCcnt).
__device__ __forceinline__ void async_copy16(unsigned lds_byte_addr, const void* gsrc) {
  asm volatile("global_load_async_to_lds_b128 %0, %1, off"
               :: "v"(lds_byte_addr), "v"((unsigned long long)gsrc)
               : "memory");
}
__device__ __forceinline__ void async_wait0() {
  asm volatile("s_wait_asynccnt 0" ::: "memory");
}

// A-matrix 16x32 bf16 fragment (ISA 7.12.2): lane r=lane&15 holds row r;
// half h=lane>>4 selects K subsets; VGPR j -> K = (j<4 ? 2j : 16+2(j-4)) + 8h.
__device__ __forceinline__ v16bf load_a_frag(const unsigned short* base, int pitch, int lane) {
  const int r = lane & 15;
  const int h = lane >> 4;
  const unsigned short* rowp = base + r * pitch + 8 * h;
  FragU f;
#pragma unroll
  for (int j = 0; j < 8; ++j) {
    const int k = (j < 4) ? (2 * j) : (16 + 2 * (j - 4));
    f.u[j] = *(const unsigned int*)(rowp + k);
  }
  return f.v;
}

// B-matrix 32x16 fragment from a K-major (K x N row-major) LDS tile:
// lane n=lane&15 holds column n; VGPR j -> K = (j<4 ? 2j : 16+2(j-4)) + 8h.
__device__ __forceinline__ v16bf load_b_frag_kmajor(const unsigned short* base, int pitch, int lane) {
  const unsigned short* p = base + (lane & 15) + 8 * (lane >> 4) * pitch;
  FragU f;
#pragma unroll
  for (int j = 0; j < 8; ++j) {
    const int k = (j < 4) ? (2 * j) : (16 + 2 * (j - 4));
    f.u[j] = (unsigned int)p[k * pitch] | ((unsigned int)p[(k + 1) * pitch] << 16);
  }
  return f.v;
}

// ---------------------------------------------------------------- convert ---
__global__ void cvt_kernel(const float* __restrict__ in,
                           unsigned short* __restrict__ out, int n) {
  for (int i = blockIdx.x * blockDim.x + threadIdx.x; i < n;
       i += gridDim.x * blockDim.x)
    out[i] = f32_to_bf16(in[i]);
}

// ------------------------------------------------------------------- GEMM ---
// C[m,n] = sum_k A[m,k] * Bw[n,k] + bias[n]
// A async-staged (double-buffered); Bw prefetched into VGPRs during compute
// and stored transposed (K-major) into the other LDS buffer.
// Block tile 128x128, BK=64, 8 waves; wave tile 64x32 (4x2 WMMA frags).
template <bool BF16OUT>
__global__ __launch_bounds__(256) void gemm_bf16_kernel(
    const unsigned short* __restrict__ A, const unsigned short* __restrict__ Bw,
    const float* __restrict__ bias, unsigned short* __restrict__ Ch,
    float* __restrict__ Cf, int K, int lda, int ldb, int ldc) {
  __shared__ unsigned short ldsA[2][128 * 72];    // M-major: 128 x (64+8)
  __shared__ unsigned short ldsBt[2][64 * 136];   // K-major: 64 x (128+8)

  const int tid  = threadIdx.x;
  const int lane = tid & 31;
  const int wid  = tid >> 5;
  const int wm   = wid >> 2;   // 0..1
  const int wn   = wid & 3;    // 0..3
  const int m0   = blockIdx.y * 128;
  const int n0   = blockIdx.x * 128;

  const v8f vz = {0.f, 0.f, 0.f, 0.f, 0.f, 0.f, 0.f, 0.f};
  v8f acc[4][2];
#pragma unroll
  for (int i = 0; i < 4; ++i)
#pragma unroll
    for (int j = 0; j < 2; ++j) acc[i][j] = vz;

  const int srow = tid >> 3;        // 0..31
  const int scol = (tid & 7) * 8;   // 0,8,...,56
  const int NT   = K >> 6;          // tiles of BK=64

  // prologue: stage tile 0 into buffer 0
#pragma unroll
  for (int p = 0; p < 4; ++p) {
    const int r = srow + p * 32;
    async_copy16((unsigned)(size_t)&ldsA[0][r * 72 + scol],
                 &A[(size_t)(m0 + r) * lda + scol]);
    union { uint4 q; unsigned short s8[8]; } tb;
    tb.q = *(const uint4*)&Bw[(size_t)(n0 + r) * ldb + scol];
#pragma unroll
    for (int i = 0; i < 8; ++i) ldsBt[0][(scol + i) * 136 + r] = tb.s8[i];
  }

  for (int it = 0; it < NT; ++it) {
    async_wait0();
    __syncthreads();

    // prefetch next tile: A via async DMA, B rows into VGPRs (stored later)
    const bool pre = (it + 1) < NT;
    const int k1 = (it + 1) << 6;
    const int nb = (it + 1) & 1;
    uint4 breg[4];
    if (pre) {
#pragma unroll
      for (int p = 0; p < 4; ++p) {
        const int r = srow + p * 32;
        async_copy16((unsigned)(size_t)&ldsA[nb][r * 72 + scol],
                     &A[(size_t)(m0 + r) * lda + k1 + scol]);
        breg[p] = *(const uint4*)&Bw[(size_t)(n0 + r) * ldb + k1 + scol];
      }
    }

    // compute on current buffer
    const unsigned short* lA  = ldsA[it & 1];
    const unsigned short* lBt = ldsBt[it & 1];
#pragma unroll
    for (int kk = 0; kk < 64; kk += 32) {
      v16bf aF[4];
#pragma unroll
      for (int i = 0; i < 4; ++i)
        aF[i] = load_a_frag(&lA[(wm * 64 + i * 16) * 72 + kk], 72, lane);
      v16bf bF[2];
#pragma unroll
      for (int j = 0; j < 2; ++j)
        bF[j] = load_b_frag_kmajor(&lBt[kk * 136 + wn * 32 + j * 16], 136, lane);
#pragma unroll
      for (int i = 0; i < 4; ++i)
#pragma unroll
        for (int j = 0; j < 2; ++j) acc[i][j] = WMMA_BF16(aF[i], bF[j], acc[i][j]);
    }

    // drain B prefetch registers into the other LDS buffer (transposed)
    if (pre) {
#pragma unroll
      for (int p = 0; p < 4; ++p) {
        const int r = srow + p * 32;
        union { uint4 q; unsigned short s8[8]; } tb;
        tb.q = breg[p];
#pragma unroll
        for (int i = 0; i < 8; ++i) ldsBt[nb][(scol + i) * 136 + r] = tb.s8[i];
      }
    }
  }

  const int nl = lane & 15;
  const int hh = lane >> 4;
#pragma unroll
  for (int j2 = 0; j2 < 2; ++j2) {
    const int n = n0 + wn * 32 + j2 * 16 + nl;
    const float bv = bias[n];
#pragma unroll
    for (int i = 0; i < 4; ++i) {
#pragma unroll
      for (int e = 0; e < 8; ++e) {
        const int m = m0 + wm * 64 + i * 16 + e + 8 * hh;
        const float val = acc[i][j2][e] + bv;
        if (BF16OUT)
          Ch[(size_t)m * ldc + n] = f32_to_bf16(val);
        else
          Cf[(size_t)m * ldc + n] = val;
      }
    }
  }
}

// -------------------------------------------------------------- attention ---
// qkvh: (B*T) x 3072 bf16, head h -> cols h*192 + [q|k|v] of 64 each.
// Computes S^T = K @ q^T so K/V chunks stream in their natural key-major
// layout (async DMA, double-buffered); q^T built once per wave in LDS with
// the 1/sqrt(dk)*log2(e) scale folded in (softmax runs in exp2 domain).
// Softmax statistics are one-query-per-lane (query = lane&15).
__global__ __launch_bounds__(256) void attn_kernel(
    const unsigned short* __restrict__ qkvh, unsigned short* __restrict__ atth) {
  __shared__ unsigned short ldsK[2][64 * 72];     // keys x dims (key-major)
  __shared__ unsigned short ldsV[2][64 * 72];     // keys x dims (key-major)
  __shared__ unsigned short ldsQ[8 * 64 * 24];    // per-wave q^T: dims x (16+8)
  __shared__ unsigned short ldsP[8 * 16 * 72];    // per-wave P: queries x (64+8)

  const int tid  = threadIdx.x;
  const int lane = tid & 31;
  const int wid  = tid >> 5;
  const int qt   = blockIdx.x & 15;   // T/128 = 16 query tiles
  const int bh   = blockIdx.x >> 4;
  const int b    = bh >> 4;
  const int head = bh & 15;
  const int q0   = qt * 128 + wid * 16;
  const int nl   = lane & 15;
  const int hh   = lane >> 4;

  // Build q^T (dims x queries) once; fold 1/sqrt(64) * log2(e) into q.
  unsigned short* qT = &ldsQ[wid * 64 * 24];
  {
    const float qscale = 0.125f * 1.44269504088896f;
    const unsigned short* qsrc =
        qkvh + (size_t)(b * 2048 + q0 + nl) * 3072 + head * 192 + hh * 32;
#pragma unroll
    for (int d = 0; d < 32; ++d)
      qT[(hh * 32 + d) * 24 + nl] = f32_to_bf16(bf16_to_f32(qsrc[d]) * qscale);
  }
  const v16bf qB0 = load_b_frag_kmajor(qT, 24, lane);            // dims 0..31
  const v16bf qB1 = load_b_frag_kmajor(qT + 32 * 24, 24, lane);  // dims 32..63

  const v8f vz = {0.f, 0.f, 0.f, 0.f, 0.f, 0.f, 0.f, 0.f};
  v8f o[4];
#pragma unroll
  for (int jn = 0; jn < 4; ++jn) o[jn] = vz;
  float mq = -3.0e38f;  // running max (log2 domain) for query (lane&15)
  float lq = 0.f;       // running sum for query (lane&15)

  const int srow = tid >> 3;
  const int scol = (tid & 7) * 8;
  unsigned short* pw = &ldsP[wid * 16 * 72];

  // prologue: stage chunk 0 into buffer 0
#pragma unroll
  for (int p = 0; p < 2; ++p) {
    const int r = srow + p * 32;
    const unsigned short* src =
        qkvh + (size_t)(b * 2048 + r) * 3072 + head * 192 + scol;
    async_copy16((unsigned)(size_t)&ldsK[0][r * 72 + scol], src + 64);
    async_copy16((unsigned)(size_t)&ldsV[0][r * 72 + scol], src + 128);
  }

  for (int ic = 0; ic < 32; ++ic) {
    async_wait0();
    __syncthreads();

    // prefetch next chunk into the other buffer while we compute
    if (ic + 1 < 32) {
      const int kc1 = (ic + 1) * 64;
      const int nb  = (ic + 1) & 1;
#pragma unroll
      for (int p = 0; p < 2; ++p) {
        const int r = srow + p * 32;
        const unsigned short* src =
            qkvh + (size_t)(b * 2048 + kc1 + r) * 3072 + head * 192 + scol;
        async_copy16((unsigned)(size_t)&ldsK[nb][r * 72 + scol], src + 64);
        async_copy16((unsigned)(size_t)&ldsV[nb][r * 72 + scol], src + 128);
      }
    }
    const unsigned short* lK = ldsK[ic & 1];
    const unsigned short* lV = ldsV[ic & 1];

    // S^T (64 keys x 16 queries): A = K rows (keys), B = q^T
    v8f s[4];
#pragma unroll
    for (int jn = 0; jn < 4; ++jn) {
      v8f sv = vz;
      sv = WMMA_BF16(load_a_frag(&lK[jn * 16 * 72], 72, lane), qB0, sv);
      sv = WMMA_BF16(load_a_frag(&lK[jn * 16 * 72 + 32], 72, lane), qB1, sv);
      s[jn] = sv;
    }

    // online softmax (exp2 domain): one query per lane; 4 parallel chains
    float pm[4], ps[4];
#pragma unroll
    for (int jn = 0; jn < 4; ++jn) {
      float m01 = fmaxf(s[jn][0], s[jn][1]);
      float m23 = fmaxf(s[jn][2], s[jn][3]);
      float m45 = fmaxf(s[jn][4], s[jn][5]);
      float m67 = fmaxf(s[jn][6], s[jn][7]);
      pm[jn] = fmaxf(fmaxf(m01, m23), fmaxf(m45, m67));
    }
    float rmax = fmaxf(fmaxf(pm[0], pm[1]), fmaxf(pm[2], pm[3]));
    rmax = fmaxf(rmax, __shfl_xor(rmax, 16));
    const float mnew  = fmaxf(mq, rmax);
    const float alpha = EXP2F(mq - mnew);
#pragma unroll
    for (int jn = 0; jn < 4; ++jn) {
      float sum = 0.f;
#pragma unroll
      for (int e = 0; e < 8; ++e) {
        const float pe = EXP2F(s[jn][e] - mnew);
        s[jn][e] = pe;
        sum += pe;
      }
      ps[jn] = sum;
    }
    float rsum = (ps[0] + ps[1]) + (ps[2] + ps[3]);
    rsum += __shfl_xor(rsum, 16);
    lq = lq * alpha + rsum;
    mq = mnew;

    // rescale O rows: row e of O is query e+8*hh -> broadcast that alpha
    float arow[8];
#pragma unroll
    for (int e = 0; e < 8; ++e) arow[e] = __shfl(alpha, e + 8 * hh, 16);
#pragma unroll
    for (int jn = 0; jn < 4; ++jn)
#pragma unroll
      for (int e = 0; e < 8; ++e) o[jn][e] *= arow[e];

    // P (queries x keys) via wave-private LDS; paired-key dword stores
#pragma unroll
    for (int jn = 0; jn < 4; ++jn)
#pragma unroll
      for (int e = 0; e < 8; e += 2)
        *(unsigned int*)&pw[nl * 72 + jn * 16 + e + 8 * hh] =
            pack_bf16x2(s[jn][e], s[jn][e + 1]);

    const v16bf p0 = load_a_frag(pw, 72, lane);        // keys 0..31
    const v16bf p1 = load_a_frag(pw + 32, 72, lane);   // keys 32..63

    // O += P @ V (V is key-major = K-major for this product)
#pragma unroll
    for (int jn = 0; jn < 4; ++jn) {
      o[jn] = WMMA_BF16(p0, load_b_frag_kmajor(&lV[jn * 16], 72, lane), o[jn]);
      o[jn] = WMMA_BF16(p1, load_b_frag_kmajor(&lV[32 * 72 + jn * 16], 72, lane), o[jn]);
    }
  }

  const float invl = 1.0f / lq;
  float irow[8];
#pragma unroll
  for (int e = 0; e < 8; ++e) irow[e] = __shfl(invl, e + 8 * hh, 16);
#pragma unroll
  for (int jn = 0; jn < 4; ++jn)
#pragma unroll
    for (int e = 0; e < 8; ++e) {
      const int trow = q0 + e + 8 * hh;
      atth[(size_t)(b * 2048 + trow) * 1024 + head * 64 + jn * 16 + nl] =
          f32_to_bf16(o[jn][e] * irow[e]);
    }
}

// --------------------------------------------------------------- launcher ---
extern "C" void kernel_launch(void* const* d_in, const int* in_sizes, int n_in,
                              void* d_out, int out_size, void* d_ws,
                              size_t ws_size, hipStream_t stream) {
  const float* x     = (const float*)d_in[0];  // (2,2048,1024)
  const float* W_qkv = (const float*)d_in[1];  // (3072,1024)
  const float* b_qkv = (const float*)d_in[2];  // (3072)
  const float* W_out = (const float*)d_in[3];  // (1024,1024)
  const float* b_out = (const float*)d_in[4];  // (1024)
  float* out = (float*)d_out;                  // (2,2048,1024) f32

  // workspace layout (bf16), 256B-aligned offsets, total 48 MiB
  char* ws = (char*)d_ws;
  unsigned short* Xh    = (unsigned short*)(ws);              //  8 MiB 4096x1024
  unsigned short* Wqkvh = (unsigned short*)(ws + 8388608);    //  6 MiB 3072x1024
  unsigned short* Wouth = (unsigned short*)(ws + 14680064);   //  2 MiB 1024x1024
  unsigned short* QKVh  = (unsigned short*)(ws + 16777216);   // 24 MiB 4096x3072
  unsigned short* Atth  = (unsigned short*)(ws + 41943040);   //  8 MiB 4096x1024

  cvt_kernel<<<2048, 256, 0, stream>>>(x, Xh, 4096 * 1024);
  cvt_kernel<<<2048, 256, 0, stream>>>(W_qkv, Wqkvh, 3072 * 1024);
  cvt_kernel<<<1024, 256, 0, stream>>>(W_out, Wouth, 1024 * 1024);

  // qkv = x @ W_qkv^T + b_qkv   (M=4096, N=3072, K=1024) -> bf16
  gemm_bf16_kernel<true><<<dim3(24, 32), 256, 0, stream>>>(
      Xh, Wqkvh, b_qkv, QKVh, nullptr, 1024, 1024, 1024, 3072);

  // flash attention per (b, head, 128-query tile): 2*16*16 = 512 blocks
  attn_kernel<<<512, 256, 0, stream>>>(QKVh, Atth);

  // out = attended @ W_out^T + b_out  (M=4096, N=1024, K=1024) -> f32
  gemm_bf16_kernel<false><<<dim3(8, 32), 256, 0, stream>>>(
      Atth, Wouth, b_out, nullptr, out, 1024, 1024, 1024, 1024);
}
